// GraphAttentionLayer_3882650435716
// MI455X (gfx1250) — compile-verified, matrix operand order checked by reading, not run on hardware
//
#include <hip/hip_runtime.h>
#include <hip/hip_bf16.h>

#define N_NODES 50000
#define N_EDGES 1600000
#define IN_DIM 128
#define OUT_DIM 32
#define HEADS 4
#define EDGE_DIM 16
#define HID (OUT_DIM * HEADS)   // 128

typedef __attribute__((ext_vector_type(16))) _Float16 v16h;
typedef __attribute__((ext_vector_type(8)))  float    v8f;

// ---- order-preserving float<->uint encoding for atomic segment-max ----
__device__ __forceinline__ unsigned enc_f32(float f) {
    unsigned b = __float_as_uint(f);
    return (b & 0x80000000u) ? ~b : (b | 0x80000000u);
}
__device__ __forceinline__ float dec_f32(unsigned u) {
    unsigned b = (u & 0x80000000u) ? (u & 0x7fffffffu) : ~u;
    return __uint_as_float(b);
}

// K0: re-initialize accumulators (must run every call: graph replay reuses ws)
__global__ __launch_bounds__(256) void gat_init_kernel(unsigned* maxv, float* denom, float* agg) {
    int i = blockIdx.x * 256 + threadIdx.x;
    if (i < N_NODES * HEADS) { maxv[i] = 0u; denom[i] = 0.0f; }
    if (i < N_NODES * HID)   { agg[i] = 0.0f; }
}

// K1: h = x @ W_lin^T via WMMA f32<-f16. One wave computes one 16x16 C tile.
// Block = 256 threads = 8 waves = the 8 N-tiles of one 16-row M-tile.
__global__ __launch_bounds__(256) void gat_gemm_kernel(const float* __restrict__ x,
                                                       const float* __restrict__ W,
                                                       float* __restrict__ h) {
    const int wave = threadIdx.x >> 5;        // 0..7 -> N tile
    const int lane = threadIdx.x & 31;
    const int m0   = blockIdx.x * 16;         // 50000/16 = 3125 tiles exactly
    const int n0   = wave * 16;
    const int rc   = lane & 15;               // row (for A) / col (for B)
    const int kgrp = (lane >> 4) * 8;         // K sub-group per half-wave

    const float* __restrict__ arow = x + (size_t)(m0 + rc) * IN_DIM + kgrp;
    const float* __restrict__ brow = W + (size_t)(n0 + rc) * IN_DIM + kgrp; // B[k][n] = W[n][k]

    v8f c = {};
    #pragma unroll
    for (int kb = 0; kb < IN_DIM; kb += 32) {
        v16h a, b;
        #pragma unroll
        for (int j = 0; j < 8; ++j) {
            a[j]     = (_Float16)arow[kb + j];
            a[8 + j] = (_Float16)arow[kb + 16 + j];
            b[j]     = (_Float16)brow[kb + j];
            b[8 + j] = (_Float16)brow[kb + 16 + j];
        }
        c = __builtin_amdgcn_wmma_f32_16x16x32_f16(false, a, false, b, (short)0, c,
                                                   false, false);
    }
    // C/D layout: VGPR r, lane l -> row m0 + r + 8*(l>=16), col n0 + (l&15)
    const int mh = (lane >> 4) << 3;
    #pragma unroll
    for (int r = 0; r < 8; ++r) {
        h[(size_t)(m0 + r + mh) * HID + n0 + rc] = c[r];
    }
}

// K1b: per-(node,head) attention terms a_src/a_dst
__global__ __launch_bounds__(256) void gat_attn_terms_kernel(const float* __restrict__ h,
                                                             const float* __restrict__ att_src,
                                                             const float* __restrict__ att_dst,
                                                             float* __restrict__ a_src,
                                                             float* __restrict__ a_dst) {
    int i = blockIdx.x * 256 + threadIdx.x;   // n*HEADS + head
    if (i >= N_NODES * HEADS) return;
    const int n  = i >> 2;
    const int hd = i & 3;
    const float* __restrict__ hv = h + (size_t)n * HID + hd * OUT_DIM;
    const float* __restrict__ as = att_src + hd * OUT_DIM;
    const float* __restrict__ ad = att_dst + hd * OUT_DIM;
    float s = 0.0f, d = 0.0f;
    #pragma unroll
    for (int j = 0; j < OUT_DIM; ++j) {
        const float v = hv[j];
        s = fmaf(v, as[j], s);
        d = fmaf(v, ad[j], d);
    }
    a_src[i] = s;
    a_dst[i] = d;
}

// K2: per-edge logits (+leaky relu), store, and atomic segment-max over dst
__global__ __launch_bounds__(256) void gat_edge_logits_kernel(const int* __restrict__ ei,
                                                              const float* __restrict__ eattr,
                                                              const float* __restrict__ Wedge,
                                                              const float* __restrict__ a_src,
                                                              const float* __restrict__ a_dst,
                                                              float* __restrict__ logits,
                                                              unsigned* __restrict__ maxv) {
    int e = blockIdx.x * 256 + threadIdx.x;
    if (e >= N_EDGES) return;
    const int s = ei[e];
    const int d = ei[N_EDGES + e];
    const float4* ep = (const float4*)(eattr + (size_t)e * EDGE_DIM);
    const float4 e0 = ep[0], e1 = ep[1], e2 = ep[2], e3 = ep[3];
    #pragma unroll
    for (int hd = 0; hd < HEADS; ++hd) {
        const float4* wp = (const float4*)(Wedge + hd * EDGE_DIM);
        const float4 w0 = wp[0], w1 = wp[1], w2 = wp[2], w3 = wp[3];
        float l = a_src[s * HEADS + hd] + a_dst[d * HEADS + hd];
        l = fmaf(e0.x, w0.x, l); l = fmaf(e0.y, w0.y, l); l = fmaf(e0.z, w0.z, l); l = fmaf(e0.w, w0.w, l);
        l = fmaf(e1.x, w1.x, l); l = fmaf(e1.y, w1.y, l); l = fmaf(e1.z, w1.z, l); l = fmaf(e1.w, w1.w, l);
        l = fmaf(e2.x, w2.x, l); l = fmaf(e2.y, w2.y, l); l = fmaf(e2.z, w2.z, l); l = fmaf(e2.w, w2.w, l);
        l = fmaf(e3.x, w3.x, l); l = fmaf(e3.y, w3.y, l); l = fmaf(e3.z, w3.z, l); l = fmaf(e3.w, w3.w, l);
        l = (l > 0.0f) ? l : 0.2f * l;                 // leaky relu, slope 0.2
        logits[(size_t)e * HEADS + hd] = l;
        atomicMax(&maxv[d * HEADS + hd], enc_f32(l));  // order-preserving uint max
    }
}

// K3: exp(logit - max[dst]) in place, atomic segment-sum into denom
__global__ __launch_bounds__(256) void gat_edge_exp_kernel(const int* __restrict__ ei,
                                                           const unsigned* __restrict__ maxv,
                                                           float* __restrict__ elog,
                                                           float* __restrict__ denom) {
    int i = blockIdx.x * 256 + threadIdx.x;   // e*HEADS + head
    if (i >= N_EDGES * HEADS) return;
    const int e  = i >> 2;
    const int hd = i & 3;
    const int d  = ei[N_EDGES + e];
    const float m  = dec_f32(maxv[d * HEADS + hd]);
    const float ex = __expf(elog[i] - m);
    elog[i] = ex;
    atomicAdd(&denom[d * HEADS + hd], ex);
}

// K4: one wave per edge; lane covers 4 of 128 dims; agg[dst] += alpha * h[src]
__global__ __launch_bounds__(256) void gat_aggregate_kernel(const int* __restrict__ ei,
                                                            const float* __restrict__ h,
                                                            const float* __restrict__ elog,
                                                            const float* __restrict__ denom,
                                                            float* __restrict__ agg) {
    const int e = blockIdx.x * 8 + (threadIdx.x >> 5);
    if (e >= N_EDGES) return;
    const int lane = threadIdx.x & 31;
    const int s  = ei[e];
    const int d  = ei[N_EDGES + e];
    const int hd = lane >> 3;                 // dims 4*lane..4*lane+3 all in head (4*lane)>>5
    const float alpha = elog[(size_t)e * HEADS + hd] /
                        (denom[d * HEADS + hd] + 1e-16f);
    const float4 hv = *(const float4*)(h + (size_t)s * HID + lane * 4);
    float* ap = agg + (size_t)d * HID + lane * 4;
    atomicAdd(ap + 0, hv.x * alpha);
    atomicAdd(ap + 1, hv.y * alpha);
    atomicAdd(ap + 2, hv.z * alpha);
    atomicAdd(ap + 3, hv.w * alpha);
}

// K5: out = mean over heads + bias
__global__ __launch_bounds__(256) void gat_finalize_kernel(const float* __restrict__ agg,
                                                           const float* __restrict__ bias,
                                                           float* __restrict__ out) {
    int i = blockIdx.x * 256 + threadIdx.x;   // n*OUT_DIM + d
    if (i >= N_NODES * OUT_DIM) return;
    const int n = i >> 5;
    const int d = i & 31;
    const float* __restrict__ a = agg + (size_t)n * HID + d;
    const float v = 0.25f * (a[0] + a[OUT_DIM] + a[2 * OUT_DIM] + a[3 * OUT_DIM]);
    out[i] = v + bias[d];
}

extern "C" void kernel_launch(void* const* d_in, const int* in_sizes, int n_in,
                              void* d_out, int out_size, void* d_ws, size_t ws_size,
                              hipStream_t stream) {
    const float* x        = (const float*)d_in[0];
    const int*   ei       = (const int*)  d_in[1];
    const float* eattr    = (const float*)d_in[2];
    const float* Wlin     = (const float*)d_in[3];
    const float* att_src  = (const float*)d_in[4];
    const float* att_dst  = (const float*)d_in[5];
    const float* bias     = (const float*)d_in[6];
    const float* Wedge    = (const float*)d_in[7];
    float* out = (float*)d_out;

    // workspace layout (floats)
    float*    ws     = (float*)d_ws;
    float*    h      = ws;                         //  6,400,000 (N*128)
    float*    a_src  = ws + 6400000;               //    200,000 (N*H)
    float*    a_dst  = ws + 6600000;               //    200,000
    unsigned* maxv   = (unsigned*)(ws + 6800000);  //    200,000
    float*    denom  = ws + 7000000;               //    200,000
    float*    elog   = ws + 7200000;               //  6,400,000 (E*H)
    float*    agg    = ws + 13600000;              //  6,400,000 (N*128)
    // total: 20,000,000 floats = 80 MB

    // K0: init accumulators
    gat_init_kernel<<<(N_NODES * HID + 255) / 256, 256, 0, stream>>>(maxv, denom, agg);

    // K1: WMMA projection h = x @ W^T  (3125 M-tiles x 8 N-tiles; 8 waves/block)
    gat_gemm_kernel<<<N_NODES / 16, 256, 0, stream>>>(x, Wlin, h);

    // K1b: per-(node,head) attention terms
    gat_attn_terms_kernel<<<(N_NODES * HEADS + 255) / 256, 256, 0, stream>>>(
        h, att_src, att_dst, a_src, a_dst);

    // K2: edge logits + segment max
    gat_edge_logits_kernel<<<(N_EDGES + 255) / 256, 256, 0, stream>>>(
        ei, eattr, Wedge, a_src, a_dst, elog, maxv);

    // K3: exp + segment sum
    gat_edge_exp_kernel<<<(N_EDGES * HEADS + 255) / 256, 256, 0, stream>>>(
        ei, maxv, elog, denom);

    // K4: weighted scatter aggregation (wave per edge)
    gat_aggregate_kernel<<<N_EDGES / 8, 256, 0, stream>>>(ei, h, elog, denom, agg);

    // K5: head mean + bias
    gat_finalize_kernel<<<(N_NODES * OUT_DIM + 255) / 256, 256, 0, stream>>>(agg, bias, out);
}
